// SentenceDifferenceEncoding_25881472926224
// MI455X (gfx1250) — matrix operand order repaired and checked by reading, not compiled, not run on hardware
//
#include <hip/hip_runtime.h>
#include <hip/hip_bf16.h>
#include <math.h>

#define LSEQ 128
#define BATCH 2048
#define DIM 300
#define KP 320      // padded D (300 -> 320)
#define H1 400      // token-weight hidden
#define KP2 416     // padded 400
#define HCONV 100
#define NCPAD 112   // conv out padded to 7 tiles
#define KD 224      // dense K padded (2 * 112)
#define NDPAD 64    // dense out padded (60 -> 64)
#define T2_TILES 16 // token tiles per wave in k_token_weight

typedef __attribute__((ext_vector_type(16))) __bf16 v16bf;
typedef __attribute__((ext_vector_type(8)))  float  v8f;
typedef __attribute__((ext_vector_type(4)))  unsigned int v4u;

union FragCvt { v16bf v; v4u u[2]; };

static __device__ __forceinline__ v8f wmma_bf16(v16bf a, v16bf b, v8f c) {
  // D = A(16x32 bf16) * B(32x16 bf16) + C(16x16 f32)
  return __builtin_amdgcn_wmma_f32_16x16x32_bf16(false, a, false, b, (short)0, c, false, false);
}

static __device__ __forceinline__ v8f zero8() {
  v8f z = {0.f,0.f,0.f,0.f,0.f,0.f,0.f,0.f};
  return z;
}

// A matrix 16xK row-major [row][k] (bf16). Lane l: row = l&15, half = l>>4;
// VGPR0-3 hold K = 8*half+0..7, VGPR4-7 hold K = 16+8*half+0..7  -> two b128 loads.
static __device__ __forceinline__ v16bf load_frag_a(const __bf16* tile, int kpad, int m0, int kb, int lane) {
  int hf = lane >> 4, row = lane & 15;
  const __bf16* p = tile + (size_t)(m0 + row) * kpad + kb;
  FragCvt f;
  f.u[0] = *(const v4u*)(p + 8 * hf);
  f.u[1] = *(const v4u*)(p + 16 + 8 * hf);
  return f.v;
}

// B matrix KxN where element B[k][n] = w[n][k], w row-major [n][kpad] (bf16).
// Lane col = l&15; VGPR0-3 hold K = 8*half+0..7, VGPR4-7 hold K = 16+8*half+0..7.
static __device__ __forceinline__ v16bf load_frag_b(const __bf16* w, int kpad, int n0, int kb, int lane) {
  int hf = lane >> 4, col = lane & 15;
  const __bf16* p = w + (size_t)(n0 + col) * kpad + kb;
  FragCvt f;
  f.u[0] = *(const v4u*)(p + 8 * hf);
  f.u[1] = *(const v4u*)(p + 16 + 8 * hf);
  return f.v;
}

// ---- CDNA5 async global->LDS copy (ASYNCcnt tracked) ----
static __device__ __forceinline__ unsigned lds_off(void* p) {
  return (unsigned)(unsigned long long)(__attribute__((address_space(3))) char*)p;
}

// nbytes must be a multiple of 16; both src and LDS dest 16B aligned.
static __device__ __forceinline__ void async_copy_g2l_b128(void* lds_dst, const void* gsrc,
                                                           unsigned nbytes, int tid, int nthreads) {
  unsigned base_off = lds_off(lds_dst);
  unsigned long long base = (unsigned long long)gsrc;
  for (unsigned c = tid; c < (nbytes >> 4); c += nthreads) {
    unsigned goff = c << 4;
    unsigned loff = base_off + goff;
    asm volatile("global_load_async_to_lds_b128 %0, %1, %2 offset:0"
                 :: "v"(loff), "v"(goff), "s"(base) : "memory");
  }
  asm volatile("s_wait_asynccnt 0" ::: "memory");
}

// ---------------- kernel 0: weights -> padded bf16 ----------------
__global__ void k_prep_weights(const float* __restrict__ W1, const float* __restrict__ W2,
                               const float* __restrict__ Wc, const float* __restrict__ Wd,
                               __bf16* __restrict__ W1b, __bf16* __restrict__ W2b,
                               __bf16* __restrict__ Wcb, __bf16* __restrict__ Wdb) {
  int i = blockIdx.x * blockDim.x + threadIdx.x;
  if (i < H1 * KP) { int n = i / KP, k = i % KP; W1b[i] = (__bf16)(k < DIM ? W1[n * DIM + k] : 0.f); }
  if (i < 32 * KP2) { int n = i / KP2, k = i % KP2; W2b[i] = (__bf16)((n < 20 && k < H1) ? W2[n * H1 + k] : 0.f); }
  if (i < NCPAD * KP) { int n = i / KP, k = i % KP; Wcb[i] = (__bf16)((n < HCONV && k < DIM) ? Wc[n * DIM + k] : 0.f); }
  if (i < NDPAD * KD) {
    int n = i / KD, k = i % KD; float v = 0.f;
    if (n < 60) {
      int kp = -1;
      if (k < 100) kp = k;
      else if (k >= NCPAD && k < NCPAD + 100) kp = 100 + (k - NCPAD);
      if (kp >= 0) v = Wd[n * 200 + kp];
    }
    Wdb[i] = (__bf16)v;
  }
}

// ---------------- kernel 1: attention + dual argmax ----------------
__global__ void __launch_bounds__(256) k_att_argmax(
    const float* __restrict__ lv, const float* __restrict__ rv,
    const int* __restrict__ llen, const int* __restrict__ rlen,
    int* __restrict__ idxR, int* __restrict__ idxL) {
  extern __shared__ __align__(16) char smem[];
  __bf16* Lt = (__bf16*)smem;                       // 128 x KP bf16
  __bf16* Rt = Lt + LSEQ * KP;                      // 128 x KP bf16
  float* redV = (float*)(smem + (size_t)2 * LSEQ * KP * 2);  // [128][8]
  int*   redI = (int*)(redV + LSEQ * 8);                     // [128][8]

  int b = blockIdx.x;
  int tid = threadIdx.x, wave = tid >> 5, lane = tid & 31;
  int hf = lane >> 4, l16 = lane & 15;

  for (int r = wave * 16; r < wave * 16 + 16; ++r) {
    const float* sL = lv + ((size_t)r * BATCH + b) * DIM;
    const float* sR = rv + ((size_t)r * BATCH + b) * DIM;
    for (int d = lane; d < KP; d += 32) {
      Lt[r * KP + d] = (__bf16)(d < DIM ? sL[d] : 0.f);
      Rt[r * KP + d] = (__bf16)(d < DIM ? sR[d] : 0.f);
    }
  }
  __syncthreads();

  int m0 = wave * 16;            // this wave's 16 rows of att
  v8f C[8];
  #pragma unroll
  for (int t = 0; t < 8; ++t) C[t] = zero8();
  for (int kb = 0; kb < KP; kb += 32) {
    v16bf a = load_frag_a(Lt, KP, m0, kb, lane);
    #pragma unroll
    for (int t = 0; t < 8; ++t) {
      v16bf bb = load_frag_b(Rt, KP, 16 * t, kb, lane);   // B = R^T
      C[t] = wmma_bf16(a, bb, C[t]);
    }
  }

  int ll = llen[b], rl = rlen[b];

  // ---- row argmax over r (idx_r) ----
  float bestv[8]; int bestn[8];
  #pragma unroll
  for (int v = 0; v < 8; ++v) { bestv[v] = -3.0e38f; bestn[v] = 0; }
  for (int t = 0; t < 8; ++t) {
    int n = 16 * t + l16;
    #pragma unroll
    for (int v = 0; v < 8; ++v) {
      int m = m0 + v + 8 * hf;
      float val = (m < ll && n < rl) ? C[t][v] : 0.0f;   // masked att (zeros participate!)
      if (val > bestv[v] || (val == bestv[v] && n < bestn[v])) { bestv[v] = val; bestn[v] = n; }
    }
  }
  for (int off = 1; off < 16; off <<= 1) {
    #pragma unroll
    for (int v = 0; v < 8; ++v) {
      float ov = __shfl_xor(bestv[v], off, 32);
      int   oi = __shfl_xor(bestn[v], off, 32);
      if (ov > bestv[v] || (ov == bestv[v] && oi < bestn[v])) { bestv[v] = ov; bestn[v] = oi; }
    }
  }
  if (l16 == 0) {
    #pragma unroll
    for (int v = 0; v < 8; ++v) {
      int m = m0 + v + 8 * hf;
      idxR[m * BATCH + b] = bestn[v];
    }
  }

  // ---- col argmax over l (idx_l) ----
  for (int t = 0; t < 8; ++t) {
    int n = 16 * t + l16;
    float cv = -3.0e38f; int cm = 0;
    #pragma unroll
    for (int v = 0; v < 8; ++v) {
      int m = m0 + v + 8 * hf;
      float val = (m < ll && n < rl) ? C[t][v] : 0.0f;
      if (val > cv) { cv = val; cm = m; }   // ascending m, strict > keeps first max
    }
    float ov = __shfl_xor(cv, 16, 32); int om = __shfl_xor(cm, 16, 32);
    if (ov > cv || (ov == cv && om < cm)) { cv = ov; cm = om; }
    if (hf == 0) { redV[n * 8 + wave] = cv; redI[n * 8 + wave] = cm; }
  }
  __syncthreads();
  if (tid < 128) {
    float cv = redV[tid * 8 + 0]; int cm = redI[tid * 8 + 0];
    for (int w2 = 1; w2 < 8; ++w2) {
      float ov = redV[tid * 8 + w2];
      if (ov > cv) { cv = ov; cm = redI[tid * 8 + w2]; }   // waves ordered by m
    }
    idxL[tid * BATCH + b] = cm;
  }
}

// ---------------- kernel 2: token-weight MLP (300->400->20, relu, max) ----------------
// W1 (bf16, 256000 B) lives in LDS for the whole block; each of 4 waves streams
// T2_TILES token tiles through private LDS scratch. 317440 B <= 320 KB WGP LDS.
__global__ void __launch_bounds__(128) k_token_weight(
    const float* __restrict__ lv, const float* __restrict__ rv,
    const __bf16* __restrict__ W1b, const __bf16* __restrict__ W2b,
    const float* __restrict__ b1, const float* __restrict__ b2,
    float* __restrict__ twL, float* __restrict__ twR) {
  extern __shared__ __align__(16) char smem[];
  int tid = threadIdx.x, wave = tid >> 5, lane = tid & 31;
  int hf = lane >> 4, l16 = lane & 15;
  __bf16* W1s = (__bf16*)smem;                                              // 400 x 320 bf16
  __bf16* Ht  = (__bf16*)(smem + (size_t)H1 * KP * 2) + (size_t)wave * 16 * KP2; // 16 x 416, Xt overlays
  __bf16* Xt  = Ht;                                                         // 16 x 320 (phase overlay)
  float*  O2  = (float*)(smem + (size_t)H1 * KP * 2 + (size_t)4 * 16 * KP2 * 2) + (size_t)wave * 16 * 32;

  // async-stage W1 into LDS once per block
  async_copy_g2l_b128(W1s, W1b, H1 * KP * 2, tid, 128);
  __syncthreads();

  for (int i = 0; i < T2_TILES; ++i) {
    int flat = (blockIdx.x * 4 + wave) * T2_TILES + i;   // 0..32767
    int side = flat >> 14;
    int rem  = flat & 16383;
    int l  = rem >> 7;                                   // rem / (BATCH/16)
    int b0 = (rem & 127) * 16;
    const float* x = (side == 0) ? lv : rv;
    float* tw = (side == 0) ? twL : twR;

    // stage 16 tokens (16 batches at fixed l) as bf16
    for (int idx = lane; idx < 16 * KP; idx += 32) {
      int row = idx / KP, d = idx - row * KP;
      float v = (d < DIM) ? x[((size_t)l * BATCH + b0 + row) * DIM + d] : 0.f;
      Xt[idx] = (__bf16)v;
    }

    // hoist A fragments (Xt dead afterwards; Ht may overwrite region)
    v16bf Af[10];
    #pragma unroll
    for (int kb = 0; kb < 10; ++kb) Af[kb] = load_frag_a(Xt, KP, 0, 32 * kb, lane);

    // zero-pad hidden columns 400..415 (re-done per tile; Xt overlay may dirty them)
    for (int idx = lane; idx < 16 * 16; idx += 32) {
      int row = idx >> 4, c = idx & 15;
      Ht[row * KP2 + H1 + c] = (__bf16)0.f;
    }

    // layer 1: B fragments from LDS-resident W1
    for (int nt = 0; nt < 25; ++nt) {
      v8f c = zero8();
      #pragma unroll
      for (int kb = 0; kb < 10; ++kb) {
        v16bf bb = load_frag_b(W1s, KP, 16 * nt, 32 * kb, lane);
        c = wmma_bf16(Af[kb], bb, c);
      }
      int n = 16 * nt + l16;
      float bias = b1[n];
      #pragma unroll
      for (int v = 0; v < 8; ++v) {
        float h = c[v] + bias;
        h = h > 0.f ? h : 0.f;
        Ht[(v + 8 * hf) * KP2 + n] = (__bf16)h;
      }
    }

    // layer 2: K = 416 (13 blocks), N tiles = 2 (20 padded to 32)
    v16bf A2[13];
    #pragma unroll
    for (int kb = 0; kb < 13; ++kb) A2[kb] = load_frag_a(Ht, KP2, 0, 32 * kb, lane);

    #pragma unroll
    for (int nt = 0; nt < 2; ++nt) {
      v8f c = zero8();
      #pragma unroll
      for (int kb = 0; kb < 13; ++kb) {
        v16bf bb = load_frag_b(W2b, KP2, 16 * nt, 32 * kb, lane);
        c = wmma_bf16(A2[kb], bb, c);
      }
      int n = 16 * nt + l16;
      float bias = (n < 20) ? b2[n] : 0.f;
      #pragma unroll
      for (int v = 0; v < 8; ++v) {
        float h = c[v] + bias;
        O2[(v + 8 * hf) * 32 + n] = h > 0.f ? h : 0.f;
      }
    }

    if (lane < 16) {
      float m = O2[lane * 32 + 0];
      for (int n = 1; n < 20; ++n) m = fmaxf(m, O2[lane * 32 + n]);
      tw[(size_t)l * BATCH + b0 + lane] = m;
    }
  }
}

// ---------------- kernel 3: diff -> conv-sum (WMMA accumulation over l) -> dense ----------------
__global__ void __launch_bounds__(128) k_diff_conv_dense(
    const float* __restrict__ lv, const float* __restrict__ rv,
    const int* __restrict__ llen, const int* __restrict__ rlen,
    const int* __restrict__ idxR, const int* __restrict__ idxL,
    const float* __restrict__ twL, const float* __restrict__ twR,
    const __bf16* __restrict__ Wcb, const float* __restrict__ bc,
    const __bf16* __restrict__ Wdb, const float* __restrict__ bd,
    float* __restrict__ out) {
  extern __shared__ __align__(16) char smem[];
  int tid = threadIdx.x, wave = tid >> 5, lane = tid & 31;
  int hf = lane >> 4, l16 = lane & 15;
  __bf16* Dt = (__bf16*)smem + (size_t)wave * 16 * KP;                      // per-wave 16x320 bf16
  float*  Fb = (float*)(smem + (size_t)4 * 16 * KP * 2);                    // [4][16][KD] f32
  __bf16* FeatsB = (__bf16*)(smem + (size_t)4 * 16 * KP * 2 + (size_t)4 * 16 * KD * 4); // 16 x KD
  __bf16* WcS = (__bf16*)(smem + (size_t)4 * 16 * KP * 2 + (size_t)4 * 16 * KD * 4 + (size_t)16 * KD * 2);

  // async-stage padded bf16 W_conv into LDS once per block
  async_copy_g2l_b128(WcS, Wcb, NCPAD * KP * 2, tid, 128);
  __syncthreads();

  int b0 = blockIdx.x * 16;
  v8f CL[7], CR[7];
  #pragma unroll
  for (int t = 0; t < 7; ++t) { CL[t] = zero8(); CR[t] = zero8(); }

  for (int l = wave; l < LSEQ; l += 4) {   // 32 uniform iterations per wave
    // --- left: |(lv - rv[idxR]) * twL| masked ---
    for (int idx = lane; idx < 16 * KP; idx += 32) {
      int row = idx / KP, d = idx - row * KP;
      int bcol = b0 + row;
      float val = 0.f;
      if (d < DIM && l < llen[bcol]) {
        int ir = idxR[l * BATCH + bcol];
        float a = lv[((size_t)l * BATCH + bcol) * DIM + d];
        float g = rv[((size_t)ir * BATCH + bcol) * DIM + d];
        val = fabsf((a - g) * twL[l * BATCH + bcol]);
      }
      Dt[idx] = (__bf16)val;
    }
    for (int kb = 0; kb < KP; kb += 32) {
      v16bf a = load_frag_a(Dt, KP, 0, kb, lane);
      #pragma unroll
      for (int t = 0; t < 7; ++t) {
        v16bf bb = load_frag_b(WcS, KP, 16 * t, kb, lane);
        CL[t] = wmma_bf16(a, bb, CL[t]);   // sum over l == C accumulation
      }
    }
    // --- right: |(rv - lv[idxL]) * twR| masked ---
    for (int idx = lane; idx < 16 * KP; idx += 32) {
      int row = idx / KP, d = idx - row * KP;
      int bcol = b0 + row;
      float val = 0.f;
      if (d < DIM && l < rlen[bcol]) {
        int il = idxL[l * BATCH + bcol];
        float a = rv[((size_t)l * BATCH + bcol) * DIM + d];
        float g = lv[((size_t)il * BATCH + bcol) * DIM + d];
        val = fabsf((a - g) * twR[l * BATCH + bcol]);
      }
      Dt[idx] = (__bf16)val;
    }
    for (int kb = 0; kb < KP; kb += 32) {
      v16bf a = load_frag_a(Dt, KP, 0, kb, lane);
      #pragma unroll
      for (int t = 0; t < 7; ++t) {
        v16bf bb = load_frag_b(WcS, KP, 16 * t, kb, lane);
        CR[t] = wmma_bf16(a, bb, CR[t]);
      }
    }
  }

  // dump per-wave partial feats, reduce across waves, add b_conv * len
  #pragma unroll
  for (int t = 0; t < 7; ++t) {
    #pragma unroll
    for (int v = 0; v < 8; ++v) {
      int row = v + 8 * hf;
      Fb[((size_t)wave * 16 + row) * KD + 16 * t + l16] = CL[t][v];
      Fb[((size_t)wave * 16 + row) * KD + NCPAD + 16 * t + l16] = CR[t][v];
    }
  }
  __syncthreads();
  for (int idx = tid; idx < 16 * KD; idx += 128) {
    int row = idx / KD, col = idx - row * KD;
    float s = Fb[idx] + Fb[16 * KD + idx] + Fb[2 * 16 * KD + idx] + Fb[3 * 16 * KD + idx];
    if (col < HCONV) s += bc[col] * (float)llen[b0 + row];
    else if (col >= NCPAD && col < NCPAD + HCONV) s += bc[col - NCPAD] * (float)rlen[b0 + row];
    FeatsB[idx] = (__bf16)s;     // pad columns stay exactly 0
  }
  __syncthreads();

  // fused dense 200->60 + relu, single wave
  if (wave == 0) {
    v8f CD[4];
    #pragma unroll
    for (int t = 0; t < 4; ++t) CD[t] = zero8();
    for (int kb = 0; kb < KD; kb += 32) {
      v16bf a = load_frag_a(FeatsB, KD, 0, kb, lane);
      #pragma unroll
      for (int t = 0; t < 4; ++t) {
        v16bf bb = load_frag_b(Wdb, KD, 16 * t, kb, lane);
        CD[t] = wmma_bf16(a, bb, CD[t]);
      }
    }
    #pragma unroll
    for (int t = 0; t < 4; ++t) {
      int n = 16 * t + l16;
      if (n < 60) {
        float bias = bd[n];
        #pragma unroll
        for (int v = 0; v < 8; ++v) {
          float h = CD[t][v] + bias;
          out[(size_t)(b0 + v + 8 * hf) * 60 + n] = h > 0.f ? h : 0.f;
        }
      }
    }
  }
}

extern "C" void kernel_launch(void* const* d_in, const int* in_sizes, int n_in,
                              void* d_out, int out_size, void* d_ws, size_t ws_size,
                              hipStream_t stream) {
  const int*   llen = (const int*)d_in[0];
  const float* lv   = (const float*)d_in[1];
  const int*   rlen = (const int*)d_in[2];
  const float* rv   = (const float*)d_in[3];
  const float* W1   = (const float*)d_in[4];
  const float* b1   = (const float*)d_in[5];
  const float* W2   = (const float*)d_in[6];
  const float* b2   = (const float*)d_in[7];
  const float* Wc   = (const float*)d_in[8];
  const float* bc   = (const float*)d_in[9];
  const float* Wd   = (const float*)d_in[10];
  const float* bd   = (const float*)d_in[11];
  float* out = (float*)d_out;

  char* ws = (char*)d_ws;
  size_t off = 0;
  int*    idxR = (int*)(ws + off);    off += (size_t)LSEQ * BATCH * 4;
  int*    idxL = (int*)(ws + off);    off += (size_t)LSEQ * BATCH * 4;
  float*  twL  = (float*)(ws + off);  off += (size_t)LSEQ * BATCH * 4;
  float*  twR  = (float*)(ws + off);  off += (size_t)LSEQ * BATCH * 4;
  __bf16* W1b  = (__bf16*)(ws + off); off += (size_t)H1 * KP * 2;
  __bf16* W2b  = (__bf16*)(ws + off); off += (size_t)32 * KP2 * 2;
  __bf16* Wcb  = (__bf16*)(ws + off); off += (size_t)NCPAD * KP * 2;
  __bf16* Wdb  = (__bf16*)(ws + off); off += (size_t)NDPAD * KD * 2;

  k_prep_weights<<<(H1 * KP + 255) / 256, 256, 0, stream>>>(W1, W2, Wc, Wd, W1b, W2b, Wcb, Wdb);

  size_t sh1 = (size_t)2 * LSEQ * KP * 2 + (size_t)LSEQ * 8 * 8;     // ~172 KB
  k_att_argmax<<<BATCH, 256, sh1, stream>>>(lv, rv, llen, rlen, idxR, idxL);

  // 512 blocks * 4 waves * 16 tiles = 32768 wave-tiles (both sides)
  size_t sh2 = (size_t)H1 * KP * 2 + (size_t)4 * 16 * KP2 * 2 + (size_t)4 * 16 * 32 * 4; // 317440 B
  k_token_weight<<<512, 128, sh2, stream>>>(lv, rv, W1b, W2b, b1, b2, twL, twR);

  size_t sh3 = (size_t)4 * 16 * KP * 2 + (size_t)4 * 16 * KD * 4 + (size_t)16 * KD * 2
             + (size_t)NCPAD * KP * 2;                                                   // 177152 B
  k_diff_conv_dense<<<BATCH / 16, 128, sh3, stream>>>(lv, rv, llen, rlen, idxR, idxL,
                                                      twL, twR, Wcb, bc, Wdb, bd, out);
}
